// EGNN_layer_10368051052943
// MI455X (gfx1250) — matrix operand order repaired and checked by reading, not compiled, not run on hardware
//
#include <hip/hip_runtime.h>

// ---------------------------------------------------------------------------
// EGNN layer for MI455X (gfx1250, wave32, WMMA).
// One wave == one node; its 16 outgoing edges (senders = arange(E) % N) form
// the M=16 dimension of bf16 16x16x32 WMMA tiles. Segment sums are in-wave
// reductions (deterministic, no atomics). All LDS fragment traffic uses
// 16-byte ds_load_b128; weights are pre-swizzled to fragment-major layout.
// The node MLP is also WMMA: the block's 4 nodes form rows 0..3 of a shared
// 16x128 tile processed by wave 0.
// ---------------------------------------------------------------------------

typedef __attribute__((ext_vector_type(16))) __bf16 v16bf;
typedef __attribute__((ext_vector_type(8)))  __bf16 v8bf;
typedef __attribute__((ext_vector_type(8)))  float  v8f;

#define Hdim 64
#define KPAD 160          // edge-MLP K padded to 5 * 32
#define EAD  16
#define WAVES 4
#define TPB  (WAVES * 32)
#define DEG  16           // edges per sender (E / N)
// LDS edge_in column order (We1 rows permuted to match at repack time):
//   0..127: h_s | h_r   128..143: edge_attr   144: dist   145..159: zero pad

__device__ __forceinline__ __bf16 f2bf(float f) { return (__bf16)f; }   // native cvt, RNE
__device__ __forceinline__ float bf2f(__bf16 b) { return (float)b; }
__device__ __forceinline__ float silu(float x) {
  return x * __builtin_amdgcn_rcpf(1.0f + __expf(-x));   // fast rcp, no IEEE div
}
__device__ __forceinline__ v8bf cvt8(float4 a, float4 b) {
  v8bf v;
  v[0] = f2bf(a.x); v[1] = f2bf(a.y); v[2] = f2bf(a.z); v[3] = f2bf(a.w);
  v[4] = f2bf(b.x); v[5] = f2bf(b.y); v[6] = f2bf(b.z); v[7] = f2bf(b.w);
  return v;
}

// A fragment (16x32, ISA 7.12.2): lane's elems 0..7 = K kb+8*half.., elems
// 8..15 = K kb+16+8*half.. -> two aligned 16B LDS loads.
__device__ __forceinline__ v16bf load_A(const __bf16* tile, int ld, int kb, int lane) {
  const __bf16* p = tile + (lane & 15) * ld + kb + 8 * (lane >> 4);
  v8bf lo = *(const v8bf*)p;
  v8bf hi = *(const v8bf*)(p + 16);
  return __builtin_shufflevector(lo, hi, 0, 1, 2, 3, 4, 5, 6, 7,
                                         8, 9, 10, 11, 12, 13, 14, 15);
}
// B fragment from fragment-major packed weights: [tile][lane][16] bf16.
__device__ __forceinline__ v16bf load_B(const __bf16* Wp, int kt, int nt, int lane) {
  const __bf16* p = Wp + (((kt << 2) + nt) * 32 + lane) * 16;
  v8bf lo = *(const v8bf*)p;
  v8bf hi = *(const v8bf*)(p + 8);
  return __builtin_shufflevector(lo, hi, 0, 1, 2, 3, 4, 5, 6, 7,
                                         8, 9, 10, 11, 12, 13, 14, 15);
}

__global__ void egnn_fused_kernel(
    const int* __restrict__ edge_index, const float* __restrict__ hg,
    const float* __restrict__ xg, const float* __restrict__ eag,
    const float* __restrict__ We1, const float* __restrict__ be1,
    const float* __restrict__ We2, const float* __restrict__ be2,
    const float* __restrict__ Wh1, const float* __restrict__ bh1,
    const float* __restrict__ Wh2, const float* __restrict__ bh2,
    const float* __restrict__ Wx1, const float* __restrict__ bx1,
    const float* __restrict__ Wx2, const float* __restrict__ bx2,
    float* __restrict__ outh, float* __restrict__ outx,
    int Nn, int Eg) {
  extern __shared__ __align__(16) char smem[];
  __bf16* We1p = (__bf16*)smem;                 // 5*4*512 fragment-major
  __bf16* We2p = We1p + 5 * 4 * 512;            // 2*4*512
  __bf16* Wx1p = We2p + 2 * 4 * 512;            // 2*4*512
  __bf16* Wh1p = Wx1p + 2 * 4 * 512;            // 4*4*512
  __bf16* Wh2p = Wh1p + 4 * 4 * 512;            // 2*4*512
  __bf16* nodeIn = Wh2p + 2 * 4 * 512;          // [16][128] (rows 0..WAVES-1 used)
  __bf16* uT   = nodeIn + 16 * 128;             // [16][64]

  const int lane = threadIdx.x & 31;
  const int wave = threadIdx.x >> 5;
  const int half = lane >> 4;
  const int nloc = lane & 15;

  __bf16* wbase = uT + 16 * Hdim + wave * (16 * KPAD + 2 * 16 * Hdim);
  __bf16* ein  = wbase;                         // [16][KPAD]
  __bf16* bufA = ein + 16 * KPAD;               // [16][H]
  __bf16* bufB = bufA + 16 * Hdim;              // [16][H]
  float* fbase = (float*)(smem +
      (size_t)(15 * 4 * 512 + 16 * 128 + 16 * Hdim +
               WAVES * (16 * KPAD + 2 * 16 * Hdim)) * 2);
  float* wf = fbase + wave * (48 + 64);
  float* xd = wf;                                // [16][3] x_i - x_j
  float* mi = wf + 48;                           // [64] summed message

  // ---- repack weights into fragment-major bf16 (shared by all waves) ----
  // entry p -> tile = p/512, lane l = (p/16)%32, elem e = p%16
  // K = kt*32 + 16*(e>>3) + 8*(l>>4) + (e&7) ; N = nt*16 + (l&15)
  for (int p = threadIdx.x; p < 5 * 4 * 512; p += TPB) {
    const int tile = p >> 9, l = (p >> 4) & 31, e = p & 15;
    const int kt = tile >> 2, nt = tile & 3;
    const int K = kt * 32 + ((e >> 3) << 4) + ((l >> 4) << 3) + (e & 7);
    const int Nc = nt * 16 + (l & 15);
    float val;
    if (K < 128)       val = We1[K * Hdim + Nc];            // h_s | h_r rows
    else if (K < 144)  val = We1[(K + 1) * Hdim + Nc];      // ea rows 129..144
    else if (K == 144) val = We1[128 * Hdim + Nc];          // dist row
    else               val = 0.0f;                          // pad rows
    We1p[p] = f2bf(val);
  }
  for (int p = threadIdx.x; p < 4 * 4 * 512; p += TPB) {
    const int tile = p >> 9, l = (p >> 4) & 31, e = p & 15;
    const int kt = tile >> 2, nt = tile & 3;
    const int K = kt * 32 + ((e >> 3) << 4) + ((l >> 4) << 3) + (e & 7);
    const int Nc = nt * 16 + (l & 15);
    Wh1p[p] = f2bf(Wh1[K * Hdim + Nc]);
    if (tile < 8) {                              // K < 64 tiles
      We2p[p] = f2bf(We2[K * Hdim + Nc]);
      Wx1p[p] = f2bf(Wx1[K * Hdim + Nc]);
      Wh2p[p] = f2bf(Wh2[K * Hdim + Nc]);
    }
  }
  __syncthreads();

  int i = blockIdx.x * WAVES + wave;
  if (i >= Nn) i = Nn - 1;                       // duplicate tail work; same values
  const int k = nloc;                            // this lane's edge row
  const long e = (long)i + (long)k * Nn;         // sender(e) == i by construction
  const int r = edge_index[(long)Eg + e];        // receiver

  // ---- gather edge_in as bf16 with 16B vector stores ----
#pragma unroll
  for (int j0 = 0; j0 < 32; j0 += 8) {
    const int c = half * 32 + j0;
    float4 s0 = *(const float4*)(hg + (long)i * Hdim + c);
    float4 s1 = *(const float4*)(hg + (long)i * Hdim + c + 4);
    *(v8bf*)(ein + k * KPAD + c) = cvt8(s0, s1);
    float4 r0 = *(const float4*)(hg + (long)r * Hdim + c);
    float4 r1 = *(const float4*)(hg + (long)r * Hdim + c + 4);
    *(v8bf*)(ein + k * KPAD + Hdim + c) = cvt8(r0, r1);
  }
  if (half == 1) {                               // zero cols 144..159 first
    v8bf z = {};
    *(v8bf*)(ein + k * KPAD + 144) = z;
    *(v8bf*)(ein + k * KPAD + 152) = z;
  }
  if (half == 0) {                               // then ea 128..143 + dist 144
    float4 e0 = *(const float4*)(eag + e * EAD);
    float4 e1 = *(const float4*)(eag + e * EAD + 4);
    float4 e2 = *(const float4*)(eag + e * EAD + 8);
    float4 e3 = *(const float4*)(eag + e * EAD + 12);
    *(v8bf*)(ein + k * KPAD + 128) = cvt8(e0, e1);
    *(v8bf*)(ein + k * KPAD + 136) = cvt8(e2, e3);
    const float dx = xg[i * 3 + 0] - xg[r * 3 + 0];
    const float dy = xg[i * 3 + 1] - xg[r * 3 + 1];
    const float dz = xg[i * 3 + 2] - xg[r * 3 + 2];
    xd[k * 3 + 0] = dx; xd[k * 3 + 1] = dy; xd[k * 3 + 2] = dz;
    ein[k * KPAD + 144] = f2bf(dx * dx + dy * dy + dz * dz);
  }
  __syncthreads();

  // ---- GEMM1: [16x160] @ We1 -> silu -> bufA (m1, bf16) ----
  {
    const v16bf a0 = load_A(ein, KPAD, 0,   lane);
    const v16bf a1 = load_A(ein, KPAD, 32,  lane);
    const v16bf a2 = load_A(ein, KPAD, 64,  lane);
    const v16bf a3 = load_A(ein, KPAD, 96,  lane);
    const v16bf a4 = load_A(ein, KPAD, 128, lane);
#pragma unroll
    for (int nt = 0; nt < 4; ++nt) {
      v8f acc = {};
      acc = __builtin_amdgcn_wmma_f32_16x16x32_bf16(false, a0, false, load_B(We1p, 0, nt, lane), (short)0, acc, false, false);
      acc = __builtin_amdgcn_wmma_f32_16x16x32_bf16(false, a1, false, load_B(We1p, 1, nt, lane), (short)0, acc, false, false);
      acc = __builtin_amdgcn_wmma_f32_16x16x32_bf16(false, a2, false, load_B(We1p, 2, nt, lane), (short)0, acc, false, false);
      acc = __builtin_amdgcn_wmma_f32_16x16x32_bf16(false, a3, false, load_B(We1p, 3, nt, lane), (short)0, acc, false, false);
      acc = __builtin_amdgcn_wmma_f32_16x16x32_bf16(false, a4, false, load_B(We1p, 4, nt, lane), (short)0, acc, false, false);
      const float bias = be1[nt * 16 + nloc];
#pragma unroll
      for (int v = 0; v < 8; ++v)
        bufA[(v + 8 * half) * Hdim + nt * 16 + nloc] = f2bf(silu(acc[v] + bias));
    }
  }
  __syncthreads();

  // ---- GEMM2: m1 @ We2 -> silu -> m_ij (bufB bf16) + column-sum -> mi ----
  {
    const v16bf a0 = load_A(bufA, Hdim, 0,  lane);
    const v16bf a1 = load_A(bufA, Hdim, 32, lane);
#pragma unroll
    for (int nt = 0; nt < 4; ++nt) {
      v8f acc = {};
      acc = __builtin_amdgcn_wmma_f32_16x16x32_bf16(false, a0, false, load_B(We2p, 0, nt, lane), (short)0, acc, false, false);
      acc = __builtin_amdgcn_wmma_f32_16x16x32_bf16(false, a1, false, load_B(We2p, 1, nt, lane), (short)0, acc, false, false);
      const float bias = be2[nt * 16 + nloc];
      float ps = 0.0f;
#pragma unroll
      for (int v = 0; v < 8; ++v) {
        const float val = silu(acc[v] + bias);
        bufB[(v + 8 * half) * Hdim + nt * 16 + nloc] = f2bf(val);
        ps += val;                               // partial column sum (8 of 16 rows)
      }
      ps += __shfl_xor(ps, 16, 32);              // combine the two M half-groups
      if (lane < 16) mi[nt * 16 + lane] = ps;    // m_i = sum_j m_ij
    }
  }
  __syncthreads();

  // ---- stage this node's row of node_in = [h_i | m_i] (row = wave) ----
  {
    const int c0 = lane * 2;
    float2 hv = *(const float2*)(hg + (long)i * Hdim + c0);
    __bf16* rowp = nodeIn + wave * 128;
    rowp[c0]     = f2bf(hv.x);
    rowp[c0 + 1] = f2bf(hv.y);
    rowp[Hdim + c0]     = f2bf(mi[c0]);
    rowp[Hdim + c0 + 1] = f2bf(mi[c0 + 1]);
  }

  // ---- GEMM3: m_ij @ Wx1 -> silu -> t (bufA bf16, reuse) ----
  {
    const v16bf a0 = load_A(bufB, Hdim, 0,  lane);
    const v16bf a1 = load_A(bufB, Hdim, 32, lane);
#pragma unroll
    for (int nt = 0; nt < 4; ++nt) {
      v8f acc = {};
      acc = __builtin_amdgcn_wmma_f32_16x16x32_bf16(false, a0, false, load_B(Wx1p, 0, nt, lane), (short)0, acc, false, false);
      acc = __builtin_amdgcn_wmma_f32_16x16x32_bf16(false, a1, false, load_B(Wx1p, 1, nt, lane), (short)0, acc, false, false);
      const float bias = bx1[nt * 16 + nloc];
#pragma unroll
      for (int v = 0; v < 8; ++v)
        bufA[(v + 8 * half) * Hdim + nt * 16 + nloc] = f2bf(silu(acc[v] + bias));
    }
  }
  __syncthreads();

  // ---- w = t @ Wx2 + bx2 ; x_new = x + (1/16) * sum_k xd_k * w_k ----
  float w = 0.0f;
  if (lane < 16) {
    float acc = bx2[0];
#pragma unroll
    for (int j0 = 0; j0 < Hdim; j0 += 8) {
      v8bf t = *(const v8bf*)(bufA + lane * Hdim + j0);
      float4 w0 = *(const float4*)(Wx2 + j0);
      float4 w1 = *(const float4*)(Wx2 + j0 + 4);
      acc += bf2f(t[0]) * w0.x + bf2f(t[1]) * w0.y + bf2f(t[2]) * w0.z + bf2f(t[3]) * w0.w;
      acc += bf2f(t[4]) * w1.x + bf2f(t[5]) * w1.y + bf2f(t[6]) * w1.z + bf2f(t[7]) * w1.w;
    }
    w = acc;
  }
  float px[3];
#pragma unroll
  for (int d = 0; d < 3; ++d)
    px[d] = (lane < 16) ? xd[lane * 3 + d] * w : 0.0f;
#pragma unroll
  for (int d = 0; d < 3; ++d)
#pragma unroll
    for (int off = 1; off < 32; off <<= 1)
      px[d] += __shfl_xor(px[d], off, 32);
  if (lane == 0) {
#pragma unroll
    for (int d = 0; d < 3; ++d)
      outx[(long)i * 3 + d] = xg[i * 3 + d] + px[d] * (1.0f / (float)DEG);
  }

  // ---- node MLP via WMMA: wave 0 processes the block's 4 nodes ----
  if (wave == 0) {
    const int blockBase = blockIdx.x * WAVES;
    const v16bf a0 = load_A(nodeIn, 128, 0,  lane);
    const v16bf a1 = load_A(nodeIn, 128, 32, lane);
    const v16bf a2 = load_A(nodeIn, 128, 64, lane);
    const v16bf a3 = load_A(nodeIn, 128, 96, lane);
#pragma unroll
    for (int nt = 0; nt < 4; ++nt) {
      v8f acc = {};
      acc = __builtin_amdgcn_wmma_f32_16x16x32_bf16(false, a0, false, load_B(Wh1p, 0, nt, lane), (short)0, acc, false, false);
      acc = __builtin_amdgcn_wmma_f32_16x16x32_bf16(false, a1, false, load_B(Wh1p, 1, nt, lane), (short)0, acc, false, false);
      acc = __builtin_amdgcn_wmma_f32_16x16x32_bf16(false, a2, false, load_B(Wh1p, 2, nt, lane), (short)0, acc, false, false);
      acc = __builtin_amdgcn_wmma_f32_16x16x32_bf16(false, a3, false, load_B(Wh1p, 3, nt, lane), (short)0, acc, false, false);
      const float bias = bh1[nt * 16 + nloc];
#pragma unroll
      for (int v = 0; v < 8; ++v)
        uT[(v + 8 * half) * Hdim + nt * 16 + nloc] = f2bf(silu(acc[v] + bias));
    }
    // same-wave LDS is in-order: uT stores are visible to the loads below
    const v16bf b0 = load_A(uT, Hdim, 0,  lane);
    const v16bf b1 = load_A(uT, Hdim, 32, lane);
#pragma unroll
    for (int nt = 0; nt < 4; ++nt) {
      v8f acc = {};
      acc = __builtin_amdgcn_wmma_f32_16x16x32_bf16(false, b0, false, load_B(Wh2p, 0, nt, lane), (short)0, acc, false, false);
      acc = __builtin_amdgcn_wmma_f32_16x16x32_bf16(false, b1, false, load_B(Wh2p, 1, nt, lane), (short)0, acc, false, false);
      if (half == 0) {                           // rows 0..3 hold real nodes
        const int col = nt * 16 + nloc;
        const float bias = bh2[col];
#pragma unroll
        for (int v = 0; v < WAVES; ++v) {
          const int iM = blockBase + v;
          if (iM < Nn)
            outh[(long)iM * Hdim + col] = hg[(long)iM * Hdim + col] + bias + acc[v];
        }
      }
    }
  }
}

extern "C" void kernel_launch(void* const* d_in, const int* in_sizes, int n_in,
                              void* d_out, int out_size, void* d_ws, size_t ws_size,
                              hipStream_t stream) {
  const int*   edge_index = (const int*)d_in[0];
  const float* h   = (const float*)d_in[1];
  const float* x   = (const float*)d_in[2];
  const float* ea  = (const float*)d_in[3];
  const float* We1 = (const float*)d_in[4];
  const float* be1 = (const float*)d_in[5];
  const float* We2 = (const float*)d_in[6];
  const float* be2 = (const float*)d_in[7];
  const float* Wh1 = (const float*)d_in[8];
  const float* bh1 = (const float*)d_in[9];
  const float* Wh2 = (const float*)d_in[10];
  const float* bh2 = (const float*)d_in[11];
  const float* Wx1 = (const float*)d_in[12];
  const float* bx1 = (const float*)d_in[13];
  const float* Wx2 = (const float*)d_in[14];
  const float* bx2 = (const float*)d_in[15];

  const int Eg = in_sizes[0] / 2;
  const int Nn = in_sizes[1] / Hdim;
  float* outh = (float*)d_out;
  float* outx = outh + (size_t)Nn * Hdim;

  const int blocks = (Nn + WAVES - 1) / WAVES;
  const size_t smem =
      (size_t)(15 * 4 * 512 + 16 * 128 + 16 * Hdim +
               WAVES * (16 * KPAD + 2 * 16 * Hdim)) * 2 +
      (size_t)WAVES * (48 + 64) * 4;

  egnn_fused_kernel<<<blocks, TPB, smem, stream>>>(
      edge_index, h, x, ea, We1, be1, We2, be2, Wh1, bh1, Wh2, bh2,
      Wx1, bx1, Wx2, bx2, outh, outx, Nn, Eg);
}
